// RobustFill_80015240724931
// MI455X (gfx1250) — matrix operand order repaired and checked by reading, not compiled, not run on hardware
//
#include <hip/hip_runtime.h>
#include <hip/hip_bf16.h>

// ---------------------------------------------------------------------------
// RobustFill forward on gfx1250 (MI455X): bf16 WMMA GEMMs, fp32 state.
// Round 3: decoder widened to 1024 threads (attention was VALU-bound),
// encoder directions merged into one launch, branch-free tanh.
// ---------------------------------------------------------------------------

#define DI static __device__ __forceinline__

typedef __attribute__((ext_vector_type(16))) __bf16 v16bf;
typedef __attribute__((ext_vector_type(8)))  float  v8f;

struct alignas(16) Q4 { unsigned int x, y, z, w; };
union ABu { v16bf v; Q4 q[2]; };

DI float b2f(unsigned short u) {
    union { float f; unsigned int i; } x; x.i = ((unsigned int)u) << 16; return x.f;
}
DI unsigned short f2b(float f) {
    union { float f; unsigned int i; } x; x.f = f;
    unsigned int r = x.i + 0x7fffu + ((x.i >> 16) & 1u);
    return (unsigned short)(r >> 16);
}
DI float sigf(float x) { return 1.f / (1.f + __expf(-x)); }
// branch-free tanh: (e^2x-1)/(e^2x+1); limits: +inf -> 1, 0 -> -1 handled.
DI float tanh_fast(float x) { float e = __expf(2.f * x); return 1.f - 2.f / (e + 1.f); }

DI v8f vzero() {
    v8f z;
#pragma unroll
    for (int i = 0; i < 8; ++i) z[i] = 0.f;
    return z;
}

// A fragment (16x32 bf16) from an LDS row-major [16][stride] buffer.
// lane<16: e0..7 <-> K kbase+0..7, e8..15 <-> K kbase+16..23
// lane>=16: K kbase+8..15 and kbase+24..31   (per CDNA5 ISA 7.12.2)
DI v16bf loadA_lds(const unsigned short* base, int stride, int kbase, int lane) {
    const unsigned short* p = base + (lane & 15) * stride + kbase + ((lane >> 4) << 3);
    ABu ab;
    ab.q[0] = *(const Q4*)(p);
    ab.q[1] = *(const Q4*)(p + 16);
    return ab.v;
}

// B fragment (32x16 bf16) from pre-swizzled global tiles [kb][nt][lane][16].
DI v16bf loadB_swz(const unsigned short* swz, int kb, int nt, int NT, int lane) {
    const unsigned short* p = swz + ((((size_t)kb * NT + nt) * 32 + lane) << 4);
    ABu ab;
    ab.q[0] = *(const Q4*)(p);
    ab.q[1] = *(const Q4*)(p + 8);
    return ab.v;
}

DI v8f wmma_bf(v16bf a, v16bf b, v8f c) {
    return __builtin_amdgcn_wmma_f32_16x16x32_bf16(false, a, false, b, (short)0, c, false, false);
}

// ---------------------------------------------------------------------------
// Prep kernels
// ---------------------------------------------------------------------------

// W[K][N] fp32 row-major -> bf16 WMMA-B tiles out[kb][nt][lane][16]
__global__ void swz_b_kernel(const float* __restrict__ W, unsigned short* __restrict__ out,
                             int K, int N) {
    size_t i = (size_t)blockIdx.x * 256 + threadIdx.x;
    if (i >= (size_t)K * N) return;
    int e = (int)(i & 15);
    int lane = (int)((i >> 4) & 31);
    size_t t = i >> 9;
    int NT = N >> 4;
    int nt = (int)(t % NT);
    int kb = (int)(t / NT);
    int n = nt * 16 + (lane & 15);
    int k = kb * 32 + ((lane >> 4) << 3) + (e & 7) + ((e >> 3) << 4);
    out[i] = f2b(W[(size_t)k * N + n]);
}

// x[l][s][0:128]=E[exs], [128:256]=E[cls]  (bf16)
__global__ void embed_kernel(const int* __restrict__ exs, const int* __restrict__ cls,
                             const float* __restrict__ E, unsigned short* __restrict__ x) {
    size_t i = (size_t)blockIdx.x * 256 + threadIdx.x;   // exact grid: 64*1024*256
    int e = (int)(i & 255);
    size_t ls = i >> 8;
    int tok = (e < 128) ? exs[ls] : cls[ls];
    x[i] = f2b(E[(size_t)tok * 128 + (e & 127)]);
}

// pre[t][g] = v_b[g] + field_emb[t] . v_Wi[:,g]   (fp32, 32x4096)
__global__ void vpre_kernel(const float* __restrict__ femb, const float* __restrict__ vWi,
                            const float* __restrict__ vb, float* __restrict__ pre) {
    int i = blockIdx.x * 256 + threadIdx.x;              // exact grid: 32*4096
    int g = i & 4095;
    int t = i >> 12;
    float s = vb[g];
    for (int e = 0; e < 1024; ++e) s += femb[t * 1024 + e] * vWi[(size_t)e * 4096 + g];
    pre[i] = s;
}

// ---------------------------------------------------------------------------
// Encoder LSTM, both directions in one launch (grid = 128 blocks):
//   dir = blockIdx.x>>6, 16 sequences per block, 64 timesteps.
// gates[16,2048] = x_t[16,256]@Wi + h[16,512]@Wh + b ; WMMA bf16.
// Per hidden tile q: compute its 4 gate tiles (acc[4] only), activate, write h
// into the *other* LDS h buffer (ping-pong) -> no post-matmul barrier needed.
// ---------------------------------------------------------------------------
__global__ __launch_bounds__(256) void enc_lstm_kernel(
    const unsigned short* __restrict__ xbf,   // [64][1024][256] bf16
    const unsigned short* __restrict__ wiFs, const unsigned short* __restrict__ whFs,
    const unsigned short* __restrict__ wiBs, const unsigned short* __restrict__ whBs,
    const float* __restrict__ biasF, const float* __restrict__ biasB,
    unsigned short* __restrict__ ctx,         // [64][1024][1024] bf16
    unsigned short* __restrict__ hTc,         // [1024][1024] bf16
    float* __restrict__ cTc) {                // [1024][1024] fp32
    __shared__ alignas(16) unsigned short sx[16 * 256];
    __shared__ alignas(16) unsigned short sh2[2][16 * 512];
    const int tid = threadIdx.x, wv = tid >> 5, lane = tid & 31;
    const int dir = blockIdx.x >> 6;
    const int s0 = (blockIdx.x & 63) * 16;
    const unsigned short* wiS = dir ? wiBs : wiFs;
    const unsigned short* whS = dir ? whBs : whFs;
    const float* bias = dir ? biasB : biasF;

    v8f creg[4];
#pragma unroll
    for (int q = 0; q < 4; ++q) creg[q] = vzero();

    float bI[4], bF[4], bG[4], bO[4];
#pragma unroll
    for (int q = 0; q < 4; ++q) {
        int n = (wv * 4 + q) * 16 + (lane & 15);
        bI[q] = bias[n]; bF[q] = bias[512 + n];
        bG[q] = bias[1024 + n]; bO[q] = bias[1536 + n];
    }
    for (int i = tid; i < 16 * 512; i += 256) sh2[0][i] = 0;

#pragma unroll 1
    for (int t = 0; t < 64; ++t) {
        const int l = dir ? (63 - t) : t;
        const unsigned short* shc = sh2[t & 1];
        unsigned short* shn = sh2[(t & 1) ^ 1];
        __syncthreads();   // prev iteration reads done; sh2[cur] complete
        for (int i = tid; i < 16 * 256; i += 256)
            sx[i] = xbf[((size_t)l * 1024 + s0 + (i >> 8)) * 256 + (i & 255)];
        __syncthreads();

#pragma unroll
        for (int q = 0; q < 4; ++q) {
            v8f acc[4];
#pragma unroll
            for (int g = 0; g < 4; ++g) acc[g] = vzero();
#pragma unroll 1
            for (int kb = 0; kb < 8; ++kb) {              // x @ Wi  (K=256)
                v16bf A = loadA_lds(sx, 256, kb * 32, lane);
#pragma unroll
                for (int g = 0; g < 4; ++g) {
                    v16bf Bm = loadB_swz(wiS, kb, g * 32 + wv * 4 + q, 128, lane);
                    acc[g] = wmma_bf(A, Bm, acc[g]);
                }
            }
#pragma unroll 1
            for (int kb = 0; kb < 16; ++kb) {             // h @ Wh  (K=512)
                v16bf A = loadA_lds(shc, 512, kb * 32, lane);
#pragma unroll
                for (int g = 0; g < 4; ++g) {
                    v16bf Bm = loadB_swz(whS, kb, g * 32 + wv * 4 + q, 128, lane);
                    acc[g] = wmma_bf(A, Bm, acc[g]);
                }
            }
            int n = (wv * 4 + q) * 16 + (lane & 15);
#pragma unroll
            for (int r = 0; r < 8; ++r) {
                int m = r + 8 * (lane >> 4);
                float iv = acc[0][r] + bI[q], fv = acc[1][r] + bF[q];
                float gv = acc[2][r] + bG[q], ov = acc[3][r] + bO[q];
                float cn = sigf(fv) * creg[q][r] + sigf(iv) * tanh_fast(gv);
                creg[q][r] = cn;
                unsigned short hb = f2b(sigf(ov) * tanh_fast(cn));
                shn[m * 512 + n] = hb;
                ctx[((size_t)l * 1024 + s0 + m) * 1024 + dir * 512 + n] = hb;
            }
        }
    }
    __syncthreads();
    // final h: t=63 wrote parity 0
    for (int i = tid; i < 16 * 512; i += 256)
        hTc[(size_t)(s0 + (i >> 9)) * 1024 + dir * 512 + (i & 511)] = sh2[0][i];
#pragma unroll
    for (int q = 0; q < 4; ++q) {
        int n = (wv * 4 + q) * 16 + (lane & 15);
#pragma unroll
        for (int r = 0; r < 8; ++r) {
            int m = r + 8 * (lane >> 4);
            cTc[(size_t)(s0 + m) * 1024 + dir * 512 + n] = creg[q][r];
        }
    }
}

// h0/c0 = max over 4 IO examples of final states
__global__ void pool_kernel(const unsigned short* __restrict__ hTc, const float* __restrict__ cTc,
                            float* __restrict__ h0, float* __restrict__ c0) {
    int i = blockIdx.x * 256 + threadIdx.x;              // exact grid: 256*1024
    int j = i & 1023, b = i >> 10;
    float hm = -3.0e38f, cm = -3.0e38f;
#pragma unroll
    for (int nio = 0; nio < 4; ++nio) {
        int s = b * 4 + nio;
        hm = fmaxf(hm, b2f(hTc[(size_t)s * 1024 + j]));
        cm = fmaxf(cm, cTc[(size_t)s * 1024 + j]);
    }
    h0[i] = hm; c0[i] = cm;
}

// ---------------------------------------------------------------------------
// Decoder: block = 16 batch rows x 32 steps, 1024 threads (32 waves).
// Attention phases are VALU-bound -> spread over 1024 lanes; WMMA phases use
// 2 column tiles per wave. Dynamic LDS ~152KB (of 320KB/WGP).
// ---------------------------------------------------------------------------
__global__ __launch_bounds__(1024) void dec_kernel(
    const float* __restrict__ h0, const float* __restrict__ c0,
    const unsigned short* __restrict__ ctx,      // [64][1024][1024] bf16
    const unsigned short* __restrict__ wAttnS,   // swz [32][64][32][16]
    const unsigned short* __restrict__ wCompS,   // swz [64][4][32][16]
    const float* __restrict__ bcomp,             // [64]
    const unsigned short* __restrict__ vWhS,     // swz [32][256][32][16]
    const float* __restrict__ vpreb,             // [32][4096]
    const int* __restrict__ actions,             // [256][32]
    float* __restrict__ out) {
    extern __shared__ char smem[];
    unsigned short* hbf  = (unsigned short*)smem;          // [16][2048]  h | ctx (bf16)
    unsigned short* qbf  = hbf + 16 * 2048;                // [16][1024]
    unsigned short* hnew = qbf + 16 * 1024;                // [16][1024]
    float* sc  = (float*)(hnew + 16 * 1024);               // [16][4][64]
    float* lg  = sc + 16 * 4 * 64;                         // [16][64]
    float* nll = lg + 16 * 64;                             // [16]

    const int tid = threadIdx.x, wv = tid >> 5, lane = tid & 31;
    const int b0 = blockIdx.x * 16;

    for (int i = tid; i < 16 * 1024; i += 1024)
        hbf[(i >> 10) * 2048 + (i & 1023)] = f2b(h0[(size_t)(b0 + (i >> 10)) * 1024 + (i & 1023)]);
    v8f creg[2];   // 2 hidden tiles per wave (32 units)
#pragma unroll
    for (int q = 0; q < 2; ++q) {
        int n = (wv * 2 + q) * 16 + (lane & 15);
#pragma unroll
        for (int r = 0; r < 8; ++r)
            creg[q][r] = c0[(size_t)(b0 + r + 8 * (lane >> 4)) * 1024 + n];
    }
    if (tid < 16) nll[tid] = 0.f;
    __syncthreads();

#pragma unroll 1
    for (int t = 0; t < 32; ++t) {
        // (a) q = h @ W_attn  -> qbf (bf16); 2 column tiles per wave
        {
            v8f acc[2];
#pragma unroll
            for (int p = 0; p < 2; ++p) acc[p] = vzero();
#pragma unroll 1
            for (int kb = 0; kb < 32; ++kb) {
                v16bf A = loadA_lds(hbf, 2048, kb * 32, lane);
#pragma unroll
                for (int p = 0; p < 2; ++p) {
                    v16bf Bm = loadB_swz(wAttnS, kb, wv * 2 + p, 64, lane);
                    acc[p] = wmma_bf(A, Bm, acc[p]);
                }
            }
#pragma unroll
            for (int p = 0; p < 2; ++p) {
                int nt = wv * 2 + p;
#pragma unroll
                for (int r = 0; r < 8; ++r)
                    qbf[(r + 8 * (lane >> 4)) * 1024 + nt * 16 + (lane & 15)] = f2b(acc[p][r]);
            }
        }
        __syncthreads();

        // (b) scores[l,b,nio] = ctx . q   (16 threads per (b,nio), 4 l's each)
        {
            int pair = tid >> 4, b = pair >> 2, nio = pair & 3, lpart = tid & 15;
            const unsigned short* qr = qbf + b * 1024;
#pragma unroll 1
            for (int li = 0; li < 4; ++li) {
                int l = lpart * 4 + li;
                const unsigned short* cr = ctx + ((size_t)l * 1024 + (size_t)(b0 + b) * 4 + nio) * 1024;
                float s = 0.f;
#pragma unroll 1
                for (int e = 0; e < 1024; e += 8) {
#pragma unroll
                    for (int i = 0; i < 8; ++i) s += b2f(cr[e + i]) * b2f(qr[e + i]);
                }
                sc[pair * 64 + l] = s;
            }
        }
        __syncthreads();

        // (c) softmax over l per (b,nio)
        if (tid < 64) {
            float* row = sc + tid * 64;
            float mx = -3.0e38f;
            for (int l = 0; l < 64; ++l) mx = fmaxf(mx, row[l]);
            float sum = 0.f;
            for (int l = 0; l < 64; ++l) { float e = __expf(row[l] - mx); row[l] = e; sum += e; }
            float inv = 1.f / sum;
            for (int l = 0; l < 64; ++l) row[l] *= inv;
        }
        __syncthreads();

        // (d) ctx_pooled = max_nio( sum_l attn*ctx ) -> hbf[:,1024:2048] (bf16)
        {
            int b = tid >> 6, eb = (tid & 63) * 16;      // 64 threads per b, 16 e's each
            float vm[16];
#pragma unroll
            for (int i = 0; i < 16; ++i) vm[i] = -3.0e38f;
#pragma unroll 1
            for (int nio = 0; nio < 4; ++nio) {
                float ac[16];
#pragma unroll
                for (int i = 0; i < 16; ++i) ac[i] = 0.f;
                const float* w = sc + (b * 4 + nio) * 64;
#pragma unroll 1
                for (int l = 0; l < 64; ++l) {
                    float wl = w[l];
                    const unsigned short* cr =
                        ctx + ((size_t)l * 1024 + (size_t)(b0 + b) * 4 + nio) * 1024 + eb;
#pragma unroll
                    for (int i = 0; i < 16; ++i) ac[i] += wl * b2f(cr[i]);
                }
#pragma unroll
                for (int i = 0; i < 16; ++i) vm[i] = fmaxf(vm[i], ac[i]);
            }
#pragma unroll
            for (int i = 0; i < 16; ++i) hbf[b * 2048 + 1024 + eb + i] = f2b(vm[i]);
        }
        __syncthreads();

        // (e) logits = [h|ctx] @ W_comp + b_comp   (waves 0..3, one 16-col tile each)
        if (wv < 4) {
            v8f acc = vzero();
#pragma unroll 1
            for (int kb = 0; kb < 64; ++kb) {
                v16bf A = loadA_lds(hbf, 2048, kb * 32, lane);
                v16bf Bm = loadB_swz(wCompS, kb, wv, 4, lane);
                acc = wmma_bf(A, Bm, acc);
            }
            float bc = bcomp[wv * 16 + (lane & 15)];
#pragma unroll
            for (int r = 0; r < 8; ++r)
                lg[(r + 8 * (lane >> 4)) * 64 + wv * 16 + (lane & 15)] = acc[r] + bc;
        }
        __syncthreads();

        // (f) nll[b] += logsumexp(logits) - logits[action]
        if (tid < 16) {
            const float* row = lg + tid * 64;
            float mx = -3.0e38f;
            for (int k = 0; k < 64; ++k) mx = fmaxf(mx, row[k]);
            float sum = 0.f;
            for (int k = 0; k < 64; ++k) sum += __expf(row[k] - mx);
            int a = actions[(size_t)(b0 + tid) * 32 + t];
            nll[tid] += (mx + __logf(sum)) - row[a];
        }

        // (g) v_lstm: gates = pre[t] + h @ v_Wh ; per hidden tile q: 4 gate
        //     tiles only (acc[4]), activate, write new h into hnew.
#pragma unroll
        for (int q = 0; q < 2; ++q) {
            v8f acc[4];
#pragma unroll
            for (int g = 0; g < 4; ++g) acc[g] = vzero();
#pragma unroll 1
            for (int kb = 0; kb < 32; ++kb) {
                v16bf A = loadA_lds(hbf, 2048, kb * 32, lane);
#pragma unroll
                for (int g = 0; g < 4; ++g) {
                    v16bf Bm = loadB_swz(vWhS, kb, g * 64 + wv * 2 + q, 256, lane);
                    acc[g] = wmma_bf(A, Bm, acc[g]);
                }
            }
            int n = (wv * 2 + q) * 16 + (lane & 15);
            float pI = vpreb[t * 4096 + n];
            float pF = vpreb[t * 4096 + 1024 + n];
            float pG = vpreb[t * 4096 + 2048 + n];
            float pO = vpreb[t * 4096 + 3072 + n];
#pragma unroll
            for (int r = 0; r < 8; ++r) {
                float iv = acc[0][r] + pI, fv = acc[1][r] + pF;
                float gv = acc[2][r] + pG, ov = acc[3][r] + pO;
                float cn = sigf(fv) * creg[q][r] + sigf(iv) * tanh_fast(gv);
                creg[q][r] = cn;
                hnew[(r + 8 * (lane >> 4)) * 1024 + n] = f2b(sigf(ov) * tanh_fast(cn));
            }
        }
        __syncthreads();                 // everyone done reading old h
        for (int i = tid; i < 16 * 1024; i += 1024)
            hbf[(i >> 10) * 2048 + (i & 1023)] = hnew[i];
        __syncthreads();
    }

    if (tid < 16) out[b0 + tid] = nll[tid];
}

// ---------------------------------------------------------------------------
// Host
// ---------------------------------------------------------------------------
extern "C" void kernel_launch(void* const* d_in, const int* in_sizes, int n_in,
                              void* d_out, int out_size, void* d_ws, size_t ws_size,
                              hipStream_t stream) {
    (void)in_sizes; (void)n_in; (void)out_size; (void)ws_size;
    const int*   exs    = (const int*)d_in[0];
    const int*   cls    = (const int*)d_in[1];
    const float* Eio    = (const float*)d_in[2];
    const float* WiF    = (const float*)d_in[3];
    const float* WhF    = (const float*)d_in[4];
    const float* bF     = (const float*)d_in[5];
    const float* WiB    = (const float*)d_in[6];
    const float* WhB    = (const float*)d_in[7];
    const float* bB     = (const float*)d_in[8];
    const float* Wattn  = (const float*)d_in[9];
    const float* Wcomp  = (const float*)d_in[10];
    const float* bcomp  = (const float*)d_in[11];
    const float* vWi    = (const float*)d_in[12];
    const float* vWh    = (const float*)d_in[13];
    const float* vb     = (const float*)d_in[14];
    const float* femb   = (const float*)d_in[15];
    const int*   acts   = (const int*)d_in[16];
    float* out = (float*)d_out;

    char* ws = (char*)d_ws;
    size_t off = 0;
    auto carve = [&](size_t bytes) -> char* {
        char* p = ws + off; off += (bytes + 255) & ~(size_t)255; return p;
    };
    unsigned short* wiFs = (unsigned short*)carve((size_t)256 * 2048 * 2);
    unsigned short* whFs = (unsigned short*)carve((size_t)512 * 2048 * 2);
    unsigned short* wiBs = (unsigned short*)carve((size_t)256 * 2048 * 2);
    unsigned short* whBs = (unsigned short*)carve((size_t)512 * 2048 * 2);
    unsigned short* wAs  = (unsigned short*)carve((size_t)1024 * 1024 * 2);
    unsigned short* wCs  = (unsigned short*)carve((size_t)2048 * 64 * 2);
    unsigned short* vWhs = (unsigned short*)carve((size_t)1024 * 4096 * 2);
    unsigned short* xbf  = (unsigned short*)carve((size_t)64 * 1024 * 256 * 2);
    unsigned short* ctx  = (unsigned short*)carve((size_t)64 * 1024 * 1024 * 2);
    unsigned short* hTc  = (unsigned short*)carve((size_t)1024 * 1024 * 2);
    float* cTc  = (float*)carve((size_t)1024 * 1024 * 4);
    float* h0   = (float*)carve((size_t)256 * 1024 * 4);
    float* c0   = (float*)carve((size_t)256 * 1024 * 4);
    float* vpre = (float*)carve((size_t)32 * 4096 * 4);

    auto swz = [&](const float* W, unsigned short* o, int K, int N) {
        size_t tot = (size_t)K * N;
        swz_b_kernel<<<(unsigned)((tot + 255) / 256), 256, 0, stream>>>(W, o, K, N);
    };
    swz(WiF, wiFs, 256, 2048);
    swz(WhF, whFs, 512, 2048);
    swz(WiB, wiBs, 256, 2048);
    swz(WhB, whBs, 512, 2048);
    swz(Wattn, wAs, 1024, 1024);
    swz(Wcomp, wCs, 2048, 64);
    swz(vWh, vWhs, 1024, 4096);

    embed_kernel<<<(64 * 1024 * 256) / 256, 256, 0, stream>>>(exs, cls, Eio, xbf);
    vpre_kernel<<<(32 * 4096) / 256, 256, 0, stream>>>(femb, vWi, vb, vpre);

    // both LSTM directions concurrently: 128 blocks (dir = blockIdx.x>>6)
    enc_lstm_kernel<<<128, 256, 0, stream>>>(xbf, wiFs, whFs, wiBs, whBs, bF, bB,
                                             ctx, hTc, cTc);

    pool_kernel<<<(256 * 1024) / 256, 256, 0, stream>>>(hTc, cTc, h0, c0);

    const int SMEM_DEC = 16 * 2048 * 2 + 16 * 1024 * 2 + 16 * 1024 * 2 +
                         16 * 4 * 64 * 4 + 16 * 64 * 4 + 16 * 4;   // 151,616 B
    (void)hipFuncSetAttribute(reinterpret_cast<const void*>(dec_kernel),
                              hipFuncAttributeMaxDynamicSharedMemorySize, SMEM_DEC);
    dec_kernel<<<16, 1024, SMEM_DEC, stream>>>(h0, c0, ctx, wAs, wCs, bcomp, vWhs, vpre, acts, out);
}
